// _MTGNNLayer_21217138442719
// MI455X (gfx1250) — matrix operand order repaired and checked by reading, not compiled, not run on hardware
//
#include <hip/hip_runtime.h>
#include <hip/hip_bf16.h>

typedef __attribute__((ext_vector_type(16))) __bf16 v16bf;
typedef __attribute__((ext_vector_type(8)))  float  v8f;

#define B_    32
#define C_    32
#define N_    512
#define T_    64
#define TO_   58
#define TP_   64
#define CSK_  64
#define ALPHA_ 0.05f

#define HTOT   (B_*C_*N_*TO_)          /* 30,408,704 */
#define PER_B  (C_*N_*TO_)             /* 950,272 */
#define HB_ELE ((size_t)B_*C_*N_*TP_)  /* 33,554,432 */

// ---------------------------------------------------------------------------
// Adjacency prep: row sums of A (+1 for self-loop) and col sums (for A^T).
// ---------------------------------------------------------------------------
__global__ void k_asum(const float* __restrict__ A, float* __restrict__ rs,
                       float* __restrict__ cs) {
    int v = blockIdx.x * blockDim.x + threadIdx.x;
    if (v >= N_) return;
    float r = 0.f, c = 0.f;
    for (int w = 0; w < N_; ++w) {
        r += A[(size_t)v * N_ + w];
        c += A[(size_t)w * N_ + v];
    }
    rs[v] = r + 1.0f;
    cs[v] = c + 1.0f;
}

__global__ void k_afill(const float* __restrict__ A, const float* __restrict__ rs,
                        const float* __restrict__ cs, __bf16* __restrict__ A1,
                        __bf16* __restrict__ A2) {
    int i = blockIdx.x * 256 + threadIdx.x;   // 512*512 elements
    int v = i >> 9, w = i & (N_ - 1);
    float d = (v == w) ? 1.0f : 0.0f;
    A1[i] = (__bf16)((A[(size_t)v * N_ + w] + d) / rs[v]);  // norm(A + I)
    A2[i] = (__bf16)((A[(size_t)w * N_ + v] + d) / cs[v]);  // norm(A^T + I)
}

// ---------------------------------------------------------------------------
// Fused gated inception TCN:  h = tanh(conv_f(x)) * sigmoid(conv_g(x))
// Writes h f32 natural (B,C,N,58) and bf16 T-MAJOR zero-padded (B*C, 64t, 512n).
// Block: (b, group of 4 nodes), 256 threads.
// ---------------------------------------------------------------------------
__global__ __launch_bounds__(256)
void k_inception(const float* __restrict__ x,
                 const float* fw0, const float* fb0, const float* fw1, const float* fb1,
                 const float* fw2, const float* fb2, const float* fw3, const float* fb3,
                 const float* gw0, const float* gb0, const float* gw1, const float* gb1,
                 const float* gw2, const float* gb2, const float* gw3, const float* gb3,
                 float* __restrict__ hf, __bf16* __restrict__ hb) {
    __shared__ float lx[4][C_][T_];           // 32 KB
    __shared__ float lfw[4608], lgw[4608];    // 36 KB
    __shared__ float lfb[32], lgb[32];
    const int KER[4]  = {2, 3, 6, 7};
    const int BOFF[4] = {0, 512, 1280, 2816};
    const float* fwp[4] = {fw0, fw1, fw2, fw3};
    const float* gwp[4] = {gw0, gw1, gw2, gw3};
    const float* fbp[4] = {fb0, fb1, fb2, fb3};
    const float* gbp[4] = {gb0, gb1, gb2, gb3};

    int tid = threadIdx.x;
    int b   = blockIdx.x >> 7;      // N/4 = 128 node-groups
    int ng  = blockIdx.x & 127;
    int n0  = ng * 4;

    #pragma unroll
    for (int it = 0; it < 8; ++it) {
        int idx = tid + it * 256;              // 0..2047 quads
        int j = idx >> 9; int c = (idx >> 4) & 31; int t4 = (idx & 15) * 4;
        *(float4*)&lx[j][c][t4] =
            *(const float4*)&x[(((size_t)b * C_ + c) * N_ + n0 + j) * T_ + t4];
    }
    #pragma unroll
    for (int bi = 0; bi < 4; ++bi) {
        int cnt = 256 * KER[bi];
        for (int idx = tid; idx < cnt; idx += 256) {
            lfw[BOFF[bi] + idx] = fwp[bi][idx];
            lgw[BOFF[bi] + idx] = gwp[bi][idx];
        }
    }
    if (tid < 32) {
        int bi = tid >> 3, cl = tid & 7;
        lfb[tid] = fbp[bi][cl];
        lgb[tid] = gbp[bi][cl];
    }
    __syncthreads();

    int j  = tid >> 6;          // node in group
    int s  = tid & 63;
    int co = s & 31;            // output channel
    int th = s >> 5;            // t-half
    int n  = n0 + j;
    int bi = co >> 3, cl = co & 7;
    int k  = KER[bi];
    int off = 7 - k;            // crop alignment: all branches end at same t
    const float* wf = &lfw[BOFF[bi] + cl * 32 * k];
    const float* wg = &lgw[BOFF[bi] + cl * 32 * k];
    float fbv = lfb[co], gbv = lgb[co];
    size_t ob  = (((size_t)b * C_ + co) * N_ + n) * (size_t)TO_;
    size_t obT = ((size_t)b * C_ + co) * (size_t)(TP_ * N_) + n;  // t-major

    for (int t = th * 29; t < th * 29 + 29; ++t) {
        float f = fbv, g = gbv;
        for (int ci = 0; ci < 32; ++ci) {
            const float* xr  = &lx[j][ci][t + off];
            const float* wfr = &wf[ci * k];
            const float* wgr = &wg[ci * k];
            for (int dt = 0; dt < k; ++dt) {
                float xv = xr[dt];
                f += xv * wfr[dt];
                g += xv * wgr[dt];
            }
        }
        float hv = tanhf(f) * (1.0f / (1.0f + expf(-g)));
        hf[ob + t]            = hv;
        hb[obT + (size_t)t * N_] = (__bf16)hv;
    }
    if (th == 1) {   // zero padding rows t = 58..63 for WMMA tiles
        for (int t = TO_; t < TP_; ++t) hb[obT + (size_t)t * N_] = (__bf16)0.0f;
    }
}

// ---------------------------------------------------------------------------
// Skip conv: out[b,cs,n] = sum_{cc,t} h[b,cc,n,t]*W[cs,cc,t] + b + skip
// Block: (b, group of 8 nodes), 256 threads (64 cs x 4 lanes, 2 nodes each).
// ---------------------------------------------------------------------------
__global__ __launch_bounds__(256)
void k_skip(const float* __restrict__ hf, const float* __restrict__ sw,
            const float* __restrict__ sb, const float* __restrict__ skin,
            float* __restrict__ out) {
    __shared__ float lh[8][C_][TO_];   // ~58 KB
    __shared__ float lw[CSK_][TO_];    // ~14.5 KB, one cc slice
    int tid = threadIdx.x;
    int b  = blockIdx.x >> 6;          // N/8 = 64 groups
    int ng = blockIdx.x & 63;
    int n0 = ng * 8;

    for (int idx = tid; idx < 8 * C_ * TO_; idx += 256) {
        int j = idx / (C_ * TO_); int r = idx % (C_ * TO_);
        int c = r / TO_; int t = r % TO_;
        lh[j][c][t] = hf[(((size_t)b * C_ + c) * N_ + n0 + j) * TO_ + t];
    }
    int cs = tid >> 2;
    int j0 = tid & 3;
    float acc0 = 0.f, acc1 = 0.f;
    for (int cc = 0; cc < C_; ++cc) {
        __syncthreads();
        for (int idx = tid; idx < CSK_ * TO_; idx += 256) {
            int c2 = idx / TO_, t2 = idx % TO_;
            lw[c2][t2] = sw[((size_t)c2 * C_ + cc) * TO_ + t2];
        }
        __syncthreads();
        for (int t = 0; t < TO_; ++t) {
            float w = lw[cs][t];
            acc0 += w * lh[j0][cc][t];
            acc1 += w * lh[j0 + 4][cc][t];
        }
    }
    float base = sb[cs];
    size_t o0 = ((size_t)b * CSK_ + cs) * N_ + n0 + j0;
    size_t o1 = o0 + 4;
    out[o0] = acc0 + base + skin[o0];
    out[o1] = acc1 + base + skin[o1];
}

// ---------------------------------------------------------------------------
// Mix-hop propagation GEMM:  Y = alpha*X0 + (1-alpha) * (A @ H)   per (b,c)
// All tensors t-major: H[(bc, t, w)] with w contiguous -> both WMMA operands
// load their fragments DIRECTLY from global with per-lane b128 (no LDS, no
// barriers).  Each wave computes TWO 16-row M-strips against shared
// B-fragments (halves L2 B traffic, 8 WMMA per 6 load-pairs).
// Block: (bc, M-tile of 128), 128 threads = 4 waves, wave = 32 rows.
// ---------------------------------------------------------------------------
__global__ __launch_bounds__(128)
void k_mixgemm(const __bf16* __restrict__ Abf, const __bf16* __restrict__ Ht,
               const __bf16* __restrict__ X0t, __bf16* __restrict__ Yt) {
    int tid  = threadIdx.x;
    int wave = tid >> 5;
    int lane = tid & 31;
    int lh   = lane & 15;
    int hi   = lane >> 4;           // half-select (ISA A/B fragment layouts)
    int bc   = blockIdx.x;
    int m0   = blockIdx.y * 128 + wave * 32;   // this wave's 32 rows

    // A-fragment: lane row = m0 + strip*16 + lh; halves 0-7 at k+hi*8, 8-15 at +16
    const __bf16* Arow0 = Abf + (size_t)(m0 + lh) * N_ + hi * 8;
    const __bf16* Arow1 = Abf + (size_t)(m0 + 16 + lh) * N_ + hi * 8;
    const __bf16* Hbase = Ht + (size_t)bc * (TP_ * N_);

    v8f acc0[4] = {};
    v8f acc1[4] = {};
    for (int kb = 0; kb < N_; kb += 32) {
        v16bf a0, a1;
        uint4* ap0 = (uint4*)&a0;
        uint4* ap1 = (uint4*)&a1;
        ap0[0] = *(const uint4*)(Arow0 + kb);        // K = kb+hi*8 .. +7
        ap0[1] = *(const uint4*)(Arow0 + kb + 16);   // K = kb+hi*8+16 .. +23
        ap1[0] = *(const uint4*)(Arow1 + kb);
        ap1[1] = *(const uint4*)(Arow1 + kb + 16);
        v16bf bm[4];
        #pragma unroll
        for (int nt = 0; nt < 4; ++nt) {
            // B-fragment: lane col t = nt*16+lh; halves 0-7 at K kb+hi*8, 8-15 at +16
            const __bf16* bp = Hbase + (size_t)(nt * 16 + lh) * N_ + kb + hi * 8;
            uint4* bq = (uint4*)&bm[nt];
            bq[0] = *(const uint4*)bp;
            bq[1] = *(const uint4*)(bp + 16);
        }
        #pragma unroll
        for (int nt = 0; nt < 4; ++nt)
            acc0[nt] = __builtin_amdgcn_wmma_f32_16x16x32_bf16(
                false, a0, false, bm[nt], (short)0, acc0[nt], false, false);
        #pragma unroll
        for (int nt = 0; nt < 4; ++nt)
            acc1[nt] = __builtin_amdgcn_wmma_f32_16x16x32_bf16(
                false, a1, false, bm[nt], (short)0, acc1[nt], false, false);
    }
    // Epilogue: D VGPR e -> node v = strip_base + hi*8 + e (consecutive),
    // t = nt*16 + lh  ->  per-lane b128 load (X0) + b128 store.
    #pragma unroll
    for (int ms = 0; ms < 2; ++ms) {
        int rb = m0 + ms * 16 + hi * 8;
        const v8f* acc = ms ? acc1 : acc0;
        #pragma unroll
        for (int nt = 0; nt < 4; ++nt) {
            int t = nt * 16 + lh;
            size_t base = (size_t)bc * (TP_ * N_) + (size_t)t * N_ + rb;
            uint4 xv = *(const uint4*)(X0t + base);
            const __bf16* xe = (const __bf16*)&xv;
            uint4 ov; __bf16* oe = (__bf16*)&ov;
            #pragma unroll
            for (int e = 0; e < 8; ++e)
                oe[e] = (__bf16)(ALPHA_ * (float)xe[e] + (1.0f - ALPHA_) * acc[nt][e]);
            *(uint4*)(Yt + base) = ov;
        }
    }
}

// ---------------------------------------------------------------------------
// Combine (1x1 convs + residual): block = (b, t), 512 threads = all nodes.
// Reads t-major bf16 tensors coalesced across lanes; 32 accumulators/thread;
// writes t-major f32 pre-LN buffer; per-batch sum/sumsq via atomics.
// ---------------------------------------------------------------------------
__global__ __launch_bounds__(512)
void k_combine(const __bf16* __restrict__ h0, const __bf16* __restrict__ h11,
               const __bf16* __restrict__ h12, const __bf16* __restrict__ h21,
               const __bf16* __restrict__ h22, const float* __restrict__ x,
               const float* __restrict__ g1w, const float* __restrict__ g1b,
               const float* __restrict__ g2w, const float* __restrict__ g2b,
               float* __restrict__ pre, float* __restrict__ stats) {
    __shared__ float swt[5][C_][33];   // padded vs bank conflicts
    __shared__ float sbias[32];
    __shared__ float red0[512], red1[512];
    int tid = threadIdx.x;             // node n
    int b = blockIdx.x / TO_;
    int t = blockIdx.x - b * TO_;

    for (int idx = tid; idx < C_ * C_; idx += 512) {
        int co = idx >> 5, c = idx & 31;
        swt[0][co][c] = g1w[co * 96 + c] + g2w[co * 96 + c];  // h0 shared term
        swt[1][co][c] = g1w[co * 96 + 32 + c];
        swt[2][co][c] = g1w[co * 96 + 64 + c];
        swt[3][co][c] = g2w[co * 96 + 32 + c];
        swt[4][co][c] = g2w[co * 96 + 64 + c];
    }
    if (tid < 32) sbias[tid] = g1b[tid] + g2b[tid];
    __syncthreads();

    int n = tid;
    float acc[32];
    #pragma unroll
    for (int co = 0; co < 32; ++co)
        acc[co] = sbias[co] + x[(((size_t)b * C_ + co) * N_ + n) * T_ + t + 6];
    for (int c = 0; c < 32; ++c) {
        size_t hbase = ((size_t)b * C_ + c) * (size_t)(TP_ * N_) + (size_t)t * N_ + n;
        float v0 = (float)h0[hbase],  v1 = (float)h11[hbase], v2 = (float)h12[hbase];
        float v3 = (float)h21[hbase], v4 = (float)h22[hbase];
        #pragma unroll
        for (int co = 0; co < 32; ++co)
            acc[co] += swt[0][co][c] * v0 + swt[1][co][c] * v1 + swt[2][co][c] * v2
                     + swt[3][co][c] * v3 + swt[4][co][c] * v4;
    }
    float lsum = 0.f, lsq = 0.f;
    #pragma unroll
    for (int co = 0; co < 32; ++co) {
        float v = acc[co];
        pre[(((size_t)b * C_ + co) * TO_ + t) * N_ + n] = v;   // coalesced
        lsum += v; lsq += v * v;
    }
    red0[tid] = lsum; red1[tid] = lsq;
    __syncthreads();
    for (int s2 = 256; s2 > 0; s2 >>= 1) {
        if (tid < s2) { red0[tid] += red0[tid + s2]; red1[tid] += red1[tid + s2]; }
        __syncthreads();
    }
    if (tid == 0) {
        atomicAdd(&stats[b * 2],     red0[0]);
        atomicAdd(&stats[b * 2 + 1], red1[0]);
    }
}

__global__ void k_zero_stats(float* stats) {
    if (threadIdx.x < 64) stats[threadIdx.x] = 0.f;
}

// ---------------------------------------------------------------------------
// LayerNorm finalize + t-major -> natural transpose; honors idx permutation.
// Block: (bc, 64-node chunk), 64 threads; per-thread register column.
// ---------------------------------------------------------------------------
__global__ __launch_bounds__(64)
void k_ln(const float* __restrict__ pre, float* __restrict__ out,
          const float* __restrict__ stats, const float* __restrict__ lnw,
          const float* __restrict__ lnb, const int* __restrict__ idx) {
    int tid = threadIdx.x;
    int bc  = blockIdx.x >> 3;                 // b*32 + c
    int n   = ((blockIdx.x & 7) << 6) + tid;
    int b = bc >> 5, c = bc & 31;
    float cnt  = (float)PER_B;
    float mean = stats[b * 2] / cnt;
    float var  = stats[b * 2 + 1] / cnt - mean * mean;
    float inv  = rsqrtf(var + 1e-5f);

    float vbuf[TO_];
    #pragma unroll
    for (int t = 0; t < TO_; ++t)                       // coalesced across lanes
        vbuf[t] = pre[((size_t)bc * TO_ + t) * N_ + n];
    int np = idx[n];
    const float* lw = &lnw[((size_t)c * N_ + np) * TO_];
    const float* lb = &lnb[((size_t)c * N_ + np) * TO_];
    float* orow = &out[((size_t)bc * N_ + n) * TO_];    // contiguous per-thread
    #pragma unroll
    for (int t = 0; t < TO_; ++t)
        orow[t] = (vbuf[t] - mean) * inv * lw[t] + lb[t];
}

// ---------------------------------------------------------------------------
extern "C" void kernel_launch(void* const* d_in, const int* in_sizes, int n_in,
                              void* d_out, int out_size, void* d_ws, size_t ws_size,
                              hipStream_t stream) {
    const float* x    = (const float*)d_in[0];
    const float* skin = (const float*)d_in[1];
    const float* A    = (const float*)d_in[2];
    const int*   idx  = (const int*)d_in[3];
    // dict order: fw_i, fb_i, gw_i, gb_i per branch
    const float* fw[4]; const float* fb[4]; const float* gw[4]; const float* gb[4];
    for (int i = 0; i < 4; ++i) {
        fw[i] = (const float*)d_in[5 + i * 4];
        fb[i] = (const float*)d_in[6 + i * 4];
        gw[i] = (const float*)d_in[7 + i * 4];
        gb[i] = (const float*)d_in[8 + i * 4];
    }
    const float* sw  = (const float*)d_in[21];
    const float* sb  = (const float*)d_in[22];
    const float* g1w = (const float*)d_in[23];
    const float* g1b = (const float*)d_in[24];
    const float* g2w = (const float*)d_in[25];
    const float* g2b = (const float*)d_in[26];
    const float* lnw = (const float*)d_in[27];
    const float* lnb = (const float*)d_in[28];

    float* out_h    = (float*)d_out;
    float* out_skip = out_h + (size_t)HTOT;

    char* ws = (char*)d_ws;
    size_t off = 0;
    float*  hf   = (float*)(ws + off);  off += (size_t)HTOT * 4;
    float*  pre  = (float*)(ws + off);  off += (size_t)HTOT * 4;
    __bf16* h0b  = (__bf16*)(ws + off); off += HB_ELE * 2;
    __bf16* h11b = (__bf16*)(ws + off); off += HB_ELE * 2;
    __bf16* h12b = (__bf16*)(ws + off); off += HB_ELE * 2;
    __bf16* h21b = (__bf16*)(ws + off); off += HB_ELE * 2;
    __bf16* h22b = (__bf16*)(ws + off); off += HB_ELE * 2;
    __bf16* A1   = (__bf16*)(ws + off); off += (size_t)N_ * N_ * 2;
    __bf16* A2   = (__bf16*)(ws + off); off += (size_t)N_ * N_ * 2;
    float*  rs   = (float*)(ws + off);  off += N_ * 4;
    float*  cs   = (float*)(ws + off);  off += N_ * 4;
    float*  stats = (float*)(ws + off); off += 64 * 4;

    k_asum <<<2, 256, 0, stream>>>(A, rs, cs);
    k_afill<<<(N_ * N_) / 256, 256, 0, stream>>>(A, rs, cs, A1, A2);

    k_inception<<<B_ * (N_ / 4), 256, 0, stream>>>(
        x, fw[0], fb[0], fw[1], fb[1], fw[2], fb[2], fw[3], fb[3],
        gw[0], gb[0], gw[1], gb[1], gw[2], gb[2], gw[3], gb[3], hf, h0b);

    k_skip<<<B_ * (N_ / 8), 256, 0, stream>>>(hf, sw, sb, skin, out_skip);

    dim3 gg(B_ * C_, N_ / 128);
    k_mixgemm<<<gg, 128, 0, stream>>>(A1, h0b,  h0b, h11b);   // h1  = aX+(1-a)A h0
    k_mixgemm<<<gg, 128, 0, stream>>>(A1, h11b, h0b, h12b);   // h2  = aX+(1-a)A h1
    k_mixgemm<<<gg, 128, 0, stream>>>(A2, h0b,  h0b, h21b);   // A^T branch
    k_mixgemm<<<gg, 128, 0, stream>>>(A2, h21b, h0b, h22b);

    k_zero_stats<<<1, 64, 0, stream>>>(stats);
    k_combine<<<B_ * TO_, 512, 0, stream>>>(h0b, h11b, h12b, h21b, h22b,
                                            x, g1w, g1b, g2w, g2b, pre, stats);
    k_ln<<<B_ * C_ * (N_ / 64), 64, 0, stream>>>(pre, out_h, stats, lnw, lnb, idx);
}
